// RelationalMemory_50182397887113
// MI455X (gfx1250) — matrix operand order repaired and checked by reading, not compiled, not run on hardware
//
#include <hip/hip_runtime.h>

// ---------------------------------------------------------------------------
// RelationalMemory on MI455X (gfx1250)
//   - bf16 WMMA GEMMs (v_wmma_f32_16x16x32_bf16), f32 accumulate
//   - weights converted+transposed to bf16 [N,K] once (L2-resident, 192MB L2)
//   - tile staging via Tensor Data Mover (tensor_load_to_lds) when available,
//     double-buffered in LDS, synced with s_wait_tensorcnt + barrier
// B=64, T=128, S=8, D=1024, H=8, DK=128
// ---------------------------------------------------------------------------

#define RB 64
#define RT 128
#define RS 8
#define RD 1024
#define RH 8
#define RDK 128

typedef __attribute__((ext_vector_type(16))) __bf16 v16bf;
typedef __attribute__((ext_vector_type(8)))  float  v8f;
typedef __attribute__((ext_vector_type(4)))  unsigned int v4u;
typedef __attribute__((ext_vector_type(8)))  int v8i;
typedef __attribute__((ext_vector_type(4)))  int v4i;

union FragBF { v16bf v; v4u q[2]; };

__device__ __forceinline__ unsigned short f2bf(float f) {
    unsigned int u = __float_as_uint(f);
    u += 0x7FFFu + ((u >> 16) & 1u);          // round to nearest even
    return (unsigned short)(u >> 16);
}
__device__ __forceinline__ float sigmoidf_(float x) {
    return 1.0f / (1.0f + __expf(-x));
}

// ------------------------- weight preparation ------------------------------
__global__ void cvt_f32_bf16(const float* __restrict__ src,
                             unsigned short* __restrict__ dst, int n) {
    int i = blockIdx.x * blockDim.x + threadIdx.x;
    if (i < n) dst[i] = f2bf(src[i]);
}

// src [K,N] f32 -> dst [N,K] bf16 (transpose once; GEMM B-tiles become
// contiguous row-major copies -> TDM-friendly, no LDS scatter)
__global__ void cvt_t_f32_bf16(const float* __restrict__ src,
                               unsigned short* __restrict__ dst, int K, int N) {
    int i = blockIdx.x * blockDim.x + threadIdx.x;
    if (i >= K * N) return;
    int n = i / K, k = i % K;
    dst[i] = f2bf(src[(size_t)k * N + n]);
}

// --------------------- per-step activation preparation ---------------------
__global__ void step_prepare(const float* __restrict__ mem,
                             const float* __restrict__ inputs, int t,
                             unsigned short* __restrict__ mem_bf,
                             unsigned short* __restrict__ tanh_bf,
                             unsigned short* __restrict__ kv_bf,
                             unsigned short* __restrict__ x_bf) {
    int i = blockIdx.x * blockDim.x + threadIdx.x;
    const int NM = RB * RS * RD;            // 512*1024
    const int NX = RB * RD;                 // 64*1024
    if (i < NM) {
        int r = i >> 10, c = i & 1023;
        float m = mem[i];
        unsigned short mb = f2bf(m);
        mem_bf[i] = mb;
        tanh_bf[i] = f2bf(tanhf(m));
        int b = r >> 3, s = r & 7;
        kv_bf[(((size_t)(b * (RS + 1) + s)) << 10) + c] = mb;
    } else if (i < NM + NX) {
        int j = i - NM;
        int b = j >> 10, c = j & 1023;
        float x = inputs[((size_t)b * RT + t) * RD + c];
        unsigned short xb = f2bf(x);
        x_bf[((size_t)b << 10) + c] = xb;
        kv_bf[(((size_t)(b * (RS + 1) + RS)) << 10) + c] = xb;
    }
}

// ------------------------------ WMMA GEMM ----------------------------------
// C[M,N] = A[M,K](bf16 row-major) * Bt[N,K](bf16 row-major, pre-transposed)
// MODE bits: 1=ReLU, 2=+Res (after ReLU), 4=write f32 Cf, 8=write bf16 Cb
#define TM 64
#define TN 128
#define TK 64
#define ASZ (TM * TK)           // 4096 ushorts
#define BSZ (TN * TK)           // 8192 ushorts
#define BUFSZ (ASZ + BSZ)       // 12288 ushorts per pipeline stage

#if __has_builtin(__builtin_amdgcn_tensor_load_to_lds) && \
    __has_builtin(__builtin_amdgcn_s_wait_tensorcnt)
#define USE_TDM 1
#else
#define USE_TDM 0
#endif

#if USE_TDM
// Issue one 2D TDM descriptor: tile rows x tkElems bf16, row stride
// strideElems, from global 'g' into LDS byte offset 'ldsByte'.
__device__ __forceinline__ void tdm_load_tile(const unsigned short* g,
                                              unsigned ldsByte, int rows,
                                              long long strideElems,
                                              int tkElems) {
    unsigned long long ga = (unsigned long long)(uintptr_t)g;
    v4u g0;
    g0[0] = 1u;                                   // count=1, user descriptor
    g0[1] = ldsByte;                              // lds_addr (bytes)
    g0[2] = (unsigned)ga;                         // global_addr[31:0]
    g0[3] = (unsigned)((ga >> 32) & 0x1FFFFFFu)   // global_addr[56:32]
            | 0x80000000u;                        // type=2 ("image")
    v8i g1;
    unsigned td0 = (unsigned)tkElems;             // tensor_dim0 == tile_dim0
    unsigned td1 = (unsigned)rows;                // tensor_dim1 == tile_dim1
    g1[0] = 0x10000;                              // data_size=1 (2 bytes)
    g1[1] = (int)((td0 & 0xFFFFu) << 16);         // tensor_dim0[15:0]
    g1[2] = (int)((td0 >> 16) | ((td1 & 0xFFFFu) << 16));
    g1[3] = (int)((td1 >> 16) | ((td0 & 0xFFFFu) << 16));   // tile_dim0
    g1[4] = (int)(td1 & 0xFFFFu);                 // tile_dim1 (tile_dim2=0)
    g1[5] = (int)(unsigned)(strideElems & 0xFFFFFFFFll);    // dim0_stride lo
    g1[6] = (int)(unsigned)((strideElems >> 32) & 0xFFFFll);// dim0_stride hi
    g1[7] = 0;                                    // dim1_stride unused (2D)
    v4i z4; z4[0] = 0; z4[1] = 0; z4[2] = 0; z4[3] = 0;
    v8i z8;
#pragma unroll
    for (int e = 0; e < 8; ++e) z8[e] = 0;
    // 6-arg form on this toolchain: (g0, g1, g2, g3, g4, cpol)
    __builtin_amdgcn_tensor_load_to_lds(g0, g1, z4, z4, z8, 0);
}
#endif

template <int MODE>
__global__ __launch_bounds__(256)
void gemm_bf16_wmma(const unsigned short* __restrict__ A,
                    const unsigned short* __restrict__ Bt,
                    const float* __restrict__ bias,
                    const float* __restrict__ Res,
                    float* __restrict__ Cf,
                    unsigned short* __restrict__ Cb,
                    int M, int N, int K) {
    extern __shared__ unsigned short smem[];   // 2 * BUFSZ (double buffer)

    const int tid  = threadIdx.x;
    const int bn   = blockIdx.x * TN;
    const int bm   = blockIdx.y * TM;
    const int wave = tid >> 5;
    const int lane = tid & 31;
    const int wm   = (wave & 1) * 32;
    const int wn   = (wave >> 1) * 32;
    const int lh   = lane >> 4;
    const int l15  = lane & 15;

    v8f acc[2][2];
#pragma unroll
    for (int a = 0; a < 2; ++a)
#pragma unroll
        for (int b = 0; b < 2; ++b)
#pragma unroll
            for (int e = 0; e < 8; ++e) acc[a][b][e] = 0.0f;

    // 8x v_wmma_f32_16x16x32_bf16 on one staged K-chunk
    auto compute_chunk = [&](const unsigned short* sa,
                             const unsigned short* sb) {
#pragma unroll
        for (int ks = 0; ks < 2; ++ks) {
            FragBF af[2], bfr[2];
#pragma unroll
            for (int fm = 0; fm < 2; ++fm) {
                int row = wm + fm * 16 + l15;
                int kb  = ks * 32 + lh * 8;
                af[fm].q[0] = *(const v4u*)(&sa[row * TK + kb]);
                af[fm].q[1] = *(const v4u*)(&sa[row * TK + kb + 16]);
            }
#pragma unroll
            for (int fn = 0; fn < 2; ++fn) {
                int col = wn + fn * 16 + l15;
                int kb  = ks * 32 + lh * 8;
                bfr[fn].q[0] = *(const v4u*)(&sb[col * TK + kb]);
                bfr[fn].q[1] = *(const v4u*)(&sb[col * TK + kb + 16]);
            }
#pragma unroll
            for (int fm = 0; fm < 2; ++fm)
#pragma unroll
                for (int fn = 0; fn < 2; ++fn)
                    acc[fm][fn] = __builtin_amdgcn_wmma_f32_16x16x32_bf16(
                        false, af[fm].v, false, bfr[fn].v,
                        (short)0, acc[fm][fn], false, false);
        }
    };

#if USE_TDM
    const int nk = K / TK;
    if (wave == 0) {    // one TDM pair per block per stage (EXEC-independent)
        tdm_load_tile(A + (size_t)bm * K, 0u, TM, K, TK);
        tdm_load_tile(Bt + (size_t)bn * K, (unsigned)(ASZ * 2), TN, K, TK);
    }
    for (int kc = 0; kc < nk; ++kc) {
        const int cb = kc & 1;
        if (wave == 0) {
            if (kc + 1 < nk) {
                const int k0n = (kc + 1) * TK;
                const unsigned base = (unsigned)((cb ^ 1) * BUFSZ * 2);
                tdm_load_tile(A + (size_t)bm * K + k0n, base, TM, K, TK);
                tdm_load_tile(Bt + (size_t)bn * K + k0n,
                              base + (unsigned)(ASZ * 2), TN, K, TK);
                __builtin_amdgcn_s_wait_tensorcnt((short)2);  // older pair done
            } else {
                __builtin_amdgcn_s_wait_tensorcnt((short)0);
            }
        }
        __syncthreads();                       // tile visible to all waves
        compute_chunk(smem + cb * BUFSZ, smem + cb * BUFSZ + ASZ);
        __syncthreads();                       // done reading before overwrite
    }
#else
    for (int k0 = 0; k0 < K; k0 += TK) {
        // A tile: 64x64 bf16, 16B chunks
#pragma unroll
        for (int it = 0; it < 2; ++it) {
            int idx = tid + it * 256;          // 0..511
            int r = idx >> 3, ck = (idx & 7) << 3;
            *(v4u*)(&smem[r * TK + ck]) =
                *(const v4u*)(A + (size_t)(bm + r) * K + k0 + ck);
        }
        // B tile: 128x64 bf16 (pre-transposed weights -> contiguous)
#pragma unroll
        for (int it = 0; it < 4; ++it) {
            int idx = tid + it * 256;          // 0..1023
            int r = idx >> 3, ck = (idx & 7) << 3;
            *(v4u*)(&smem[ASZ + r * TK + ck]) =
                *(const v4u*)(Bt + (size_t)(bn + r) * K + k0 + ck);
        }
        __syncthreads();
        compute_chunk(smem, smem + ASZ);
        __syncthreads();
    }
#endif

    // ---- epilogue: bias (+ReLU) (+residual), f32 / bf16 outputs ----
    constexpr bool RELU = (MODE & 1) != 0;
    constexpr bool RES  = (MODE & 2) != 0;
    constexpr bool CF   = (MODE & 4) != 0;
    constexpr bool CB   = (MODE & 8) != 0;
#pragma unroll
    for (int fm = 0; fm < 2; ++fm)
#pragma unroll
        for (int fn = 0; fn < 2; ++fn) {
            int col = bn + wn + fn * 16 + l15;
            float bv = bias[col];
#pragma unroll
            for (int e = 0; e < 8; ++e) {
                int row = bm + wm + fm * 16 + lh * 8 + e;
                float x = acc[fm][fn][e] + bv;
                if constexpr (RELU) x = fmaxf(x, 0.0f);
                size_t o = (size_t)row * N + col;
                if constexpr (RES) x += Res[o];
                if constexpr (CF) Cf[o] = x;
                if constexpr (CB) Cb[o] = f2bf(x);
            }
        }
}

// ---------------------------- attention core -------------------------------
__global__ __launch_bounds__(128)
void attention_kernel(const float* __restrict__ Q,
                      const float* __restrict__ Kp,
                      const float* __restrict__ Vp,
                      unsigned short* __restrict__ attn_bf) {
    __shared__ float ks[(RS + 1) * RDK];
    __shared__ float vs[(RS + 1) * RDK];
    __shared__ float sc[RS * (RS + 1)];
    __shared__ float pr[RS * (RS + 1)];

    const int bh = blockIdx.x;
    const int b = bh / RH, h = bh % RH;
    const int tid = threadIdx.x;
    const float scale = 0.08838834764831845f;   // 1/sqrt(128)

    for (int idx = tid; idx < (RS + 1) * RDK; idx += 128) {
        int s = idx >> 7, d = idx & 127;
        size_t o = ((size_t)(b * (RS + 1) + s)) * RD + h * RDK + d;
        ks[idx] = Kp[o];
        vs[idx] = Vp[o];
    }
    __syncthreads();

    if (tid < RS * (RS + 1)) {
        int sq = tid / (RS + 1), sk = tid % (RS + 1);
        const float* q = Q + ((size_t)(b * RS + sq)) * RD + h * RDK;
        float acc = 0.0f;
        for (int d = 0; d < RDK; ++d) acc += q[d] * ks[sk * RDK + d];
        sc[tid] = acc * scale;
    }
    __syncthreads();

    if (tid < RS) {
        float mx = -1e30f;
        for (int j = 0; j <= RS; ++j) mx = fmaxf(mx, sc[tid * (RS + 1) + j]);
        float sum = 0.0f;
        float e[RS + 1];
        for (int j = 0; j <= RS; ++j) {
            e[j] = __expf(sc[tid * (RS + 1) + j] - mx);
            sum += e[j];
        }
        float inv = 1.0f / sum;
        for (int j = 0; j <= RS; ++j) pr[tid * (RS + 1) + j] = e[j] * inv;
    }
    __syncthreads();

    int d = tid;   // 0..127
    for (int sq = 0; sq < RS; ++sq) {
        float acc = 0.0f;
#pragma unroll
        for (int sk = 0; sk <= RS; ++sk)
            acc += pr[sq * (RS + 1) + sk] * vs[sk * RDK + d];
        attn_bf[((size_t)(b * RS + sq)) * RD + h * RDK + d] = f2bf(acc);
    }
}

// ------------------------------- gating ------------------------------------
__global__ void finalize_kernel(const float* __restrict__ gx,   // [64,2048]
                                const float* __restrict__ gu,   // [512,2048]
                                const float* __restrict__ mlp,  // [512,1024]
                                const float* __restrict__ mem,  // [512,1024]
                                float* __restrict__ out) {      // [512,1024]
    int i = blockIdx.x * blockDim.x + threadIdx.x;
    if (i >= RB * RS * RD) return;
    int r = i >> 10, c = i & 1023, b = r >> 3;
    float gi = sigmoidf_(gx[(size_t)b * 2 * RD + c] + gu[(size_t)r * 2 * RD + c]);
    float gf = sigmoidf_(gx[(size_t)b * 2 * RD + c + RD] +
                         gu[(size_t)r * 2 * RD + c + RD]);
    out[i] = gi * tanhf(mlp[i]) + gf * mem[i];
}

// ---------------------------------------------------------------------------
template <int MODE>
static inline void gemm_go(const unsigned short* A, const unsigned short* Bt,
                           const float* bias, const float* Res, float* Cf,
                           unsigned short* Cb, int M, int N, int K,
                           hipStream_t s) {
    dim3 g(N / TN, M / TM);
    gemm_bf16_wmma<MODE><<<g, 256, 2 * BUFSZ * 2, s>>>(A, Bt, bias, Res, Cf,
                                                       Cb, M, N, K);
}

extern "C" void kernel_launch(void* const* d_in, const int* in_sizes, int n_in,
                              void* d_out, int out_size, void* d_ws, size_t ws_size,
                              hipStream_t stream) {
    (void)in_sizes; (void)n_in; (void)out_size; (void)ws_size;

    const float* inputs = (const float*)d_in[0];   // [B,T,D]
    const float* mem_in = (const float*)d_in[1];   // [B,S,D]
    const float* Wq = (const float*)d_in[2];  const float* bq = (const float*)d_in[3];
    const float* Wk = (const float*)d_in[4];  const float* bk = (const float*)d_in[5];
    const float* Wv = (const float*)d_in[6];  const float* bv = (const float*)d_in[7];
    const float* Wo = (const float*)d_in[8];  const float* bo = (const float*)d_in[9];
    const float* Wm1 = (const float*)d_in[10]; const float* bm1 = (const float*)d_in[11];
    const float* Wm2 = (const float*)d_in[12]; const float* bm2 = (const float*)d_in[13];
    const float* Ww = (const float*)d_in[14]; const float* bw = (const float*)d_in[15];
    const float* Wu = (const float*)d_in[16]; const float* bu = (const float*)d_in[17];

    // ---- workspace bump allocator ----
    char* p = (char*)d_ws;
    auto alloc = [&](size_t bytes) -> void* {
        void* r = (void*)p;
        p += (bytes + 255) & ~(size_t)255;
        return r;
    };
    const size_t DD = (size_t)RD * RD;                 // 1M
    const size_t MROWS = (size_t)RB * RS;              // 512
    const size_t KVROWS = (size_t)RB * (RS + 1);       // 576

    // bf16 weights, pre-transposed to [N,K]
    unsigned short* Wq_t  = (unsigned short*)alloc(DD * 2);
    unsigned short* Wk_t  = (unsigned short*)alloc(DD * 2);
    unsigned short* Wv_t  = (unsigned short*)alloc(DD * 2);
    unsigned short* Wo_t  = (unsigned short*)alloc(DD * 2);
    unsigned short* Wm1_t = (unsigned short*)alloc(DD * 2);
    unsigned short* Wm2_t = (unsigned short*)alloc(DD * 2);
    unsigned short* Ww_t  = (unsigned short*)alloc(2 * DD * 2);
    unsigned short* Wu_t  = (unsigned short*)alloc(2 * DD * 2);

    float* memA = (float*)alloc(MROWS * RD * 4);
    float* memB = (float*)alloc(MROWS * RD * 4);
    unsigned short* mem_bf  = (unsigned short*)alloc(MROWS * RD * 2);
    unsigned short* tanh_bf = (unsigned short*)alloc(MROWS * RD * 2);
    unsigned short* kv_bf   = (unsigned short*)alloc(KVROWS * RD * 2);
    unsigned short* x_bf    = (unsigned short*)alloc((size_t)RB * RD * 2);
    float* Qf = (float*)alloc(MROWS * RD * 4);
    float* Kf = (float*)alloc(KVROWS * RD * 4);
    float* Vf = (float*)alloc(KVROWS * RD * 4);
    unsigned short* attn_bf = (unsigned short*)alloc(MROWS * RD * 2);
    float* na_f  = (float*)alloc(MROWS * RD * 4);
    unsigned short* na_bf  = (unsigned short*)alloc(MROWS * RD * 2);
    unsigned short* hid_bf = (unsigned short*)alloc(MROWS * RD * 2);
    float* mlp_f = (float*)alloc(MROWS * RD * 4);
    float* gx = (float*)alloc((size_t)RB * 2 * RD * 4);
    float* gu = (float*)alloc(MROWS * 2 * RD * 4);

    // ---- one-time weight convert+transpose to bf16 [N,K] ----
    auto cvt_t = [&](const float* s, unsigned short* d, int N) {
        int n = RD * N;
        cvt_t_f32_bf16<<<(n + 255) / 256, 256, 0, stream>>>(s, d, RD, N);
    };
    cvt_t(Wq, Wq_t, RD);   cvt_t(Wk, Wk_t, RD);
    cvt_t(Wv, Wv_t, RD);   cvt_t(Wo, Wo_t, RD);
    cvt_t(Wm1, Wm1_t, RD); cvt_t(Wm2, Wm2_t, RD);
    cvt_t(Ww, Ww_t, 2 * RD); cvt_t(Wu, Wu_t, 2 * RD);

    float* bufs[2] = {memA, memB};
    const int NPREP = (int)((MROWS + RB) * RD);        // 589824

    for (int t = 0; t < RT; ++t) {
        const float* cur = (t == 0) ? mem_in : bufs[t & 1];
        float* nxt = (t == RT - 1) ? (float*)d_out : bufs[(t + 1) & 1];

        step_prepare<<<(NPREP + 255) / 256, 256, 0, stream>>>(
            cur, inputs, t, mem_bf, tanh_bf, kv_bf, x_bf);

        // q/k/v projections (f32 out only -> MODE=4)
        gemm_go<4>(mem_bf, Wq_t, bq, nullptr, Qf, nullptr,
                   (int)MROWS, RD, RD, stream);
        gemm_go<4>(kv_bf, Wk_t, bk, nullptr, Kf, nullptr,
                   (int)KVROWS, RD, RD, stream);
        gemm_go<4>(kv_bf, Wv_t, bv, nullptr, Vf, nullptr,
                   (int)KVROWS, RD, RD, stream);

        attention_kernel<<<RB * RH, 128, 0, stream>>>(Qf, Kf, Vf, attn_bf);

        // next_attn = mem + attn@Wo + bo   (res + f32 + bf16 -> MODE=14)
        gemm_go<14>(attn_bf, Wo_t, bo, cur, na_f, na_bf,
                    (int)MROWS, RD, RD, stream);
        // hidden = relu(next_attn@Wm1 + bm1)   (relu + bf16 -> MODE=9)
        gemm_go<9>(na_bf, Wm1_t, bm1, nullptr, nullptr, hid_bf,
                   (int)MROWS, RD, RD, stream);
        // next_mlp = next_attn + relu(hidden@Wm2 + bm2)  (relu+res+f32 -> 7)
        gemm_go<7>(hid_bf, Wm2_t, bm2, na_f, mlp_f, nullptr,
                   (int)MROWS, RD, RD, stream);
        // gate projections (f32 out -> MODE=4)
        gemm_go<4>(x_bf, Ww_t, bw, nullptr, gx, nullptr,
                   RB, 2 * RD, RD, stream);
        gemm_go<4>(tanh_bf, Wu_t, bu, nullptr, gu, nullptr,
                   (int)MROWS, 2 * RD, RD, stream);

        finalize_kernel<<<(RB * RS * RD) / 256, 256, 0, stream>>>(
            gx, gu, mlp_f, cur, nxt);
    }
}
